// MultiQueryAttention_73693048864852
// MI455X (gfx1250) — matrix operand order repaired and checked by reading, not compiled, not run on hardware
//
#include <hip/hip_runtime.h>

// ---------------------------------------------------------------------------
// CDNA5 (gfx1250) Multi-Query Attention, bf16 WMMA pipeline + TDM staging.
//   q = x @ w_q^T ; k = x @ w_k^T ; v = x @ w_v^T
//   att = softmax(q k^T / sqrt(128)) ; o = att v ; out = o @ w_out^T
// GEMMs: v_wmma_f32_16x16x32_bf16 (f32 accumulate).
// Attention K/V tiles: tensor_load_to_lds (Tensor Data Mover), double-buffered,
// synchronized with s_wait_tensorcnt.
// ---------------------------------------------------------------------------

typedef __bf16 bf16_t;
typedef __attribute__((ext_vector_type(16))) __bf16 v16bf;
typedef __attribute__((ext_vector_type(8)))  float  v8f;
typedef __attribute__((ext_vector_type(4)))  __bf16 bf16x4;
typedef __attribute__((ext_vector_type(4)))  unsigned int u32x4;
typedef __attribute__((ext_vector_type(8)))  int i32x8;
typedef __attribute__((ext_vector_type(4)))  int i32x4;

union FragBF { v16bf v; uint4 q[2]; };
union FragF8 { v8f  v; float f[8]; };

__device__ __forceinline__ v8f wmma_bf16(v16bf a, v16bf b, v8f c) {
  // (neg_a, A, neg_b, B, c_mod, C, reuse_a, reuse_b)
  return __builtin_amdgcn_wmma_f32_16x16x32_bf16(false, a, false, b,
                                                 (short)0, c, false, false);
}

__device__ __forceinline__ bf16x4 load4_bf16(const float* p) {
  float4 f = *(const float4*)p;
  bf16x4 r = { (__bf16)f.x, (__bf16)f.y, (__bf16)f.z, (__bf16)f.w };
  return r;
}
__device__ __forceinline__ bf16x4 load4_bf16(const __bf16* p) {
  return *(const bf16x4*)p;
}
__device__ __forceinline__ void store_out(__bf16* p, float v) { *p = (__bf16)v; }
__device__ __forceinline__ void store_out(float*  p, float v) { *p = v; }

// ---------------------------------------------------------------------------
// Tensor Data Mover: 2D bf16 tile load, Global -> LDS.
// Descriptor layout per CDNA5 ISA ch.8 (group0 128b, group1 256b).
//   tensor_d0 = contiguous length, tensor_d1 = #rows, stride0 = row stride,
//   tile_d0 x tile_d1 tile starting at gptr, written contiguously at lds_off.
// This toolchain exposes the 6-arg builtin:
//   (u32x4 g0, i32x8 g1, i32x4, i32x4, i32x8, i32 cpol)
// ---------------------------------------------------------------------------
__device__ __forceinline__ void tdm_load_2d_bf16(const void* gptr, unsigned lds_off,
                                                 unsigned tensor_d0, unsigned tensor_d1,
                                                 unsigned tile_d0, unsigned tile_d1,
                                                 unsigned stride0)
{
  unsigned long long ga = (unsigned long long)(size_t)gptr;
  u32x4 g0 = { 1u,                                    // count=1 (valid), load
               lds_off,                               // lds_addr (bytes)
               (unsigned)ga,                          // global_addr[31:0]
               (unsigned)((ga >> 32) & 0x01FFFFFFull) // global_addr[56:32]
                   | (2u << 30) };                    // type=2 ("image")
  i32x8 g1 = { (int)(1u << 16),                       // data_size=1 -> 2 bytes
               (int)((tensor_d0 & 0xFFFFu) << 16),    // tensor_dim0 lo16
               (int)((tensor_d0 >> 16) | ((tensor_d1 & 0xFFFFu) << 16)),
               (int)((tensor_d1 >> 16) | (tile_d0 << 16)),
               (int)tile_d1,                          // tile_dim1; tile_dim2=0
               (int)stride0,                          // tensor_dim0_stride lo32
               0, 0 };                                // stride0 hi / dim1_stride
  i32x4 z4 = { 0, 0, 0, 0 };
  i32x8 z8 = { 0, 0, 0, 0, 0, 0, 0, 0 };
  __builtin_amdgcn_tensor_load_to_lds(g0, g1, z4, z4, z8, 0);
}

// ---------------------------------------------------------------------------
// Tiled GEMM:  C[M,N] = A[M,K] * W[N,K]^T   (W is [out,in] like nn.Linear)
// A: fp32 or bf16 (converted to bf16 in LDS).  Block = 256 thr = 8 waves.
// Tile: M=128 (16 rows/wave), N=64 (4 wmma col-tiles), K step = 32.
// Requires M%128==0, N%64==0, K%32==0 (true for all uses here).
// ---------------------------------------------------------------------------
template <typename AT, typename OT>
__global__ void __launch_bounds__(256)
gemm_wmma(const AT* __restrict__ A, const float* __restrict__ W,
          OT* __restrict__ C, int M, int N, int K)
{
  __shared__ __bf16 Abuf[128][32];  // 8 KB
  __shared__ __bf16 Bbuf[64][32];   // 4 KB

  const int tid  = threadIdx.x;
  const int wave = tid >> 5;
  const int lane = tid & 31;
  const int half = lane >> 4;
  const int lrow = lane & 15;

  const int rowbase = blockIdx.y * 128;
  const int nbase   = blockIdx.x * 64;

  FragF8 acc[4];
#pragma unroll
  for (int nt = 0; nt < 4; ++nt) {
    v8f z = {0.f, 0.f, 0.f, 0.f, 0.f, 0.f, 0.f, 0.f};
    acc[nt].v = z;
  }

  const int nk = K >> 5;
  for (int kt = 0; kt < nk; ++kt) {
    const int k0 = kt << 5;
    // ---- stage A tile 128x32 (cvt -> bf16) ----
#pragma unroll
    for (int p = 0; p < 4; ++p) {
      int idx = (p * 256 + tid) << 2;           // element in tile
      int r = idx >> 5, c = idx & 31;
      *(bf16x4*)&Abuf[r][c] =
          load4_bf16(A + (size_t)(rowbase + r) * K + k0 + c);
    }
    // ---- stage B tile 64x32 (rows of W = columns of B) ----
#pragma unroll
    for (int p = 0; p < 2; ++p) {
      int idx = (p * 256 + tid) << 2;
      int r = idx >> 5, c = idx & 31;
      *(bf16x4*)&Bbuf[r][c] =
          load4_bf16(W + (size_t)(nbase + r) * K + k0 + c);
    }
    // prefetch next K tile (global_prefetch_b8)
    if (kt + 1 < nk) {
      __builtin_prefetch(A + (size_t)(rowbase + (tid >> 3)) * K + k0 + 32 + ((tid & 7) << 2), 0, 0);
      __builtin_prefetch(W + (size_t)(nbase + (tid >> 2) % 64) * K + k0 + 32, 0, 0);
    }
    __syncthreads();

    // ---- A fragment: rows [wave*16 .. +15], K 0..31 of this step ----
    FragBF fa;
    {
      const __bf16* ap = &Abuf[wave * 16 + lrow][half * 8];
      fa.q[0] = *(const uint4*)ap;
      fa.q[1] = *(const uint4*)(ap + 16);
    }
#pragma unroll
    for (int nt = 0; nt < 4; ++nt) {
      FragBF fb;
      const __bf16* bp = &Bbuf[nt * 16 + lrow][half * 8];
      fb.q[0] = *(const uint4*)bp;
      fb.q[1] = *(const uint4*)(bp + 16);
      acc[nt].v = wmma_bf16(fa.v, fb.v, acc[nt].v);
    }
    __syncthreads();
  }

  // ---- epilogue: C-fragment (lane = col, VGPR j -> row j + 8*half) ----
#pragma unroll
  for (int nt = 0; nt < 4; ++nt) {
    int coln = nbase + nt * 16 + lrow;
#pragma unroll
    for (int j = 0; j < 8; ++j) {
      int rowm = rowbase + wave * 16 + half * 8 + j;
      store_out(&C[(size_t)rowm * N + coln], acc[nt].f[j]);
    }
  }
}

// ---------------------------------------------------------------------------
// Transpose v [B*S, 128] -> vT [B, 128, S]  (bf16)
// ---------------------------------------------------------------------------
__global__ void transpose_v(const __bf16* __restrict__ vb,
                            __bf16* __restrict__ vtb, int S)
{
  int idx = blockIdx.x * 256 + threadIdx.x;   // over B*S*128
  int d = idx & 127;
  int s = (idx >> 7) % S;
  int b = (idx >> 7) / S;
  vtb[((size_t)b * 128 + d) * S + s] = vb[idx];
}

// ---------------------------------------------------------------------------
// Flash-style MQA attention. One KV head shared by 16 Q heads.
// Block = 128 thr (4 waves); each wave owns 16 query rows.
// grid = (S/64, HEADS, B).  qb/aob: [B*S, 2048] bf16;
// kb: [B*S, 128] bf16; vtb: [B, 128, S] bf16.
// K/V^T tiles staged by the Tensor Data Mover, double-buffered: wave 0 issues
// tile i+1 while all waves compute on tile i; s_wait_tensorcnt gates reuse.
// ---------------------------------------------------------------------------
__global__ void __launch_bounds__(128)
mqa_attention(const __bf16* __restrict__ qb, const __bf16* __restrict__ kb,
              const __bf16* __restrict__ vtb, __bf16* __restrict__ aob, int S)
{
  constexpr int D = 128, H = 2048;
  __shared__ __bf16 Kbuf[2][64][128];   // [buf][t_local][d]   32 KB
  __shared__ __bf16 Vbuf[2][128][64];   // [buf][d][t_local]   32 KB
  __shared__ __bf16 Pbuf[4][16][64];    // per-wave P           8 KB

  const int b = blockIdx.z, h = blockIdx.y;
  const int tid  = threadIdx.x;
  const int wave = tid >> 5;
  const int lane = tid & 31;
  const int half = lane >> 4;
  const int lrow = lane & 15;
  const int row0 = blockIdx.x * 64 + wave * 16;

  const __bf16* kbase = kb + (size_t)b * S * D;
  const __bf16* vbase = vtb + (size_t)b * D * S;

  // ---- Q fragments for this wave's 16 rows: 4 K-steps of 32 (D=128) ----
  FragBF fq[4];
  {
    const __bf16* qrow = qb + ((size_t)b * S + row0 + lrow) * H + h * D;
#pragma unroll
    for (int ks = 0; ks < 4; ++ks) {
      const __bf16* p = qrow + ks * 32 + half * 8;
      fq[ks].q[0] = *(const uint4*)p;
      fq[ks].q[1] = *(const uint4*)(p + 16);
    }
  }

  float mstat[8], lstat[8];
  FragF8 oacc[8];
#pragma unroll
  for (int j = 0; j < 8; ++j) { mstat[j] = -1e30f; lstat[j] = 0.f; }
#pragma unroll
  for (int n = 0; n < 8; ++n) {
    v8f z = {0.f, 0.f, 0.f, 0.f, 0.f, 0.f, 0.f, 0.f};
    oacc[n].v = z;
  }

  const float scale = 0.08838834764831845f;  // 1/sqrt(128)
  const int nT = S >> 6;                     // 64-token tiles

  // ---- TDM prologue: issue tile 0 into buffer 0 (wave 0 only) ----
  if (wave == 0) {
    // K tile: tensor [S x 128], tile [64 rows x 128], row stride 128
    tdm_load_2d_bf16(kbase, (unsigned)(size_t)&Kbuf[0][0][0],
                     (unsigned)D, (unsigned)S, (unsigned)D, 64u, (unsigned)D);
    // V^T tile: tensor [128 x S], tile [128 rows x 64], row stride S
    tdm_load_2d_bf16(vbase, (unsigned)(size_t)&Vbuf[0][0][0],
                     (unsigned)S, (unsigned)D, 64u, (unsigned)D, (unsigned)S);
  }

  for (int it = 0; it < nT; ++it) {
    const int cur = it & 1;
    if (wave == 0) {
      if (it + 1 < nT) {
        const int nxt = cur ^ 1;
        const int t1 = (it + 1) << 6;
        tdm_load_2d_bf16(kbase + (size_t)t1 * D,
                         (unsigned)(size_t)&Kbuf[nxt][0][0],
                         (unsigned)D, (unsigned)S, (unsigned)D, 64u, (unsigned)D);
        tdm_load_2d_bf16(vbase + t1,
                         (unsigned)(size_t)&Vbuf[nxt][0][0],
                         (unsigned)S, (unsigned)D, 64u, (unsigned)D, (unsigned)S);
        __builtin_amdgcn_s_wait_tensorcnt(2);  // tile `it` (K+V) complete
      } else {
        __builtin_amdgcn_s_wait_tensorcnt(0);  // last tile complete
      }
    }
    __syncthreads();   // publish TDM-written LDS to all waves

    // ---- scores S = Q K^T (4 col tiles of 16 tokens) ----
    FragF8 sc[4];
#pragma unroll
    for (int nt = 0; nt < 4; ++nt) {
      v8f c = {0.f, 0.f, 0.f, 0.f, 0.f, 0.f, 0.f, 0.f};
#pragma unroll
      for (int ks = 0; ks < 4; ++ks) {
        FragBF fb;
        const __bf16* bp = &Kbuf[cur][nt * 16 + lrow][ks * 32 + half * 8];
        fb.q[0] = *(const uint4*)bp;
        fb.q[1] = *(const uint4*)(bp + 16);
        c = wmma_bf16(fq[ks].v, fb.v, c);
      }
      sc[nt].v = c;
    }

    // ---- online softmax: rows m = half*8 + j, cols across 16-lane half ----
    float rmax[8];
#pragma unroll
    for (int j = 0; j < 8; ++j) {
      float v0 = sc[0].f[j] * scale, v1 = sc[1].f[j] * scale;
      float v2 = sc[2].f[j] * scale, v3 = sc[3].f[j] * scale;
      sc[0].f[j] = v0; sc[1].f[j] = v1; sc[2].f[j] = v2; sc[3].f[j] = v3;
      float r = fmaxf(fmaxf(v0, v1), fmaxf(v2, v3));
#pragma unroll
      for (int msk = 1; msk < 16; msk <<= 1)
        r = fmaxf(r, __shfl_xor(r, msk, 32));
      rmax[j] = r;
    }

    float fscale[8];
#pragma unroll
    for (int j = 0; j < 8; ++j) {
      float mnew = fmaxf(mstat[j], rmax[j]);
      fscale[j] = __expf(mstat[j] - mnew);
      mstat[j] = mnew;
      float s = 0.f;
#pragma unroll
      for (int nt = 0; nt < 4; ++nt) {
        float p = __expf(sc[nt].f[j] - mnew);
        sc[nt].f[j] = p;
        s += p;
      }
#pragma unroll
      for (int msk = 1; msk < 16; msk <<= 1)
        s += __shfl_xor(s, msk, 32);
      lstat[j] = lstat[j] * fscale[j] + s;
    }

    // rescale running output
#pragma unroll
    for (int n = 0; n < 8; ++n)
#pragma unroll
      for (int j = 0; j < 8; ++j) oacc[n].f[j] *= fscale[j];

    // ---- stage P (C-layout -> A-layout via per-wave LDS buffer) ----
#pragma unroll
    for (int nt = 0; nt < 4; ++nt)
#pragma unroll
      for (int j = 0; j < 8; ++j)
        Pbuf[wave][half * 8 + j][nt * 16 + lrow] = (__bf16)sc[nt].f[j];
    // wave-private buffer: in-order DS ops, compiler inserts s_wait_dscnt

    FragBF fp[2];
#pragma unroll
    for (int ks = 0; ks < 2; ++ks) {
      const __bf16* pp = &Pbuf[wave][lrow][ks * 32 + half * 8];
      fp[ks].q[0] = *(const uint4*)pp;
      fp[ks].q[1] = *(const uint4*)(pp + 16);
    }

    // ---- O += P V  (8 output d-tiles, K = 64 tokens = 2 steps) ----
#pragma unroll
    for (int n = 0; n < 8; ++n) {
      v8f c = oacc[n].v;
#pragma unroll
      for (int ks = 0; ks < 2; ++ks) {
        FragBF fb;
        const __bf16* bp = &Vbuf[cur][n * 16 + lrow][ks * 32 + half * 8];
        fb.q[0] = *(const uint4*)bp;
        fb.q[1] = *(const uint4*)(bp + 16);
        c = wmma_bf16(fp[ks].v, fb.v, c);
      }
      oacc[n].v = c;
    }
    __syncthreads();   // all waves done with buf `cur` before it is re-filled
  }

  // ---- finalize: divide by l, write bf16 head slice ----
  float linv[8];
#pragma unroll
  for (int j = 0; j < 8; ++j) linv[j] = 1.0f / lstat[j];
#pragma unroll
  for (int n = 0; n < 8; ++n) {
    int coldst = h * D + n * 16 + lrow;
#pragma unroll
    for (int j = 0; j < 8; ++j) {
      size_t r = (size_t)b * S + row0 + half * 8 + j;
      aob[r * H + coldst] = (__bf16)(oacc[n].f[j] * linv[j]);
    }
  }
}

// ---------------------------------------------------------------------------
// Host-side launcher
// ---------------------------------------------------------------------------
extern "C" void kernel_launch(void* const* d_in, const int* in_sizes, int n_in,
                              void* d_out, int out_size, void* d_ws, size_t ws_size,
                              hipStream_t stream)
{
  (void)in_sizes; (void)n_in; (void)out_size; (void)ws_size;
  constexpr int Bsz = 2, S = 2048, H = 2048, D = 128, HEADS = 16;
  const int M = Bsz * S;  // 4096 (batch folded)

  const float* x     = (const float*)d_in[0];
  const float* w_q   = (const float*)d_in[1];
  const float* w_k   = (const float*)d_in[2];
  const float* w_v   = (const float*)d_in[3];
  const float* w_out = (const float*)d_in[4];
  float* out = (float*)d_out;

  char* ws = (char*)d_ws;
  __bf16* qbuf = (__bf16*)ws;                                    // M*H
  __bf16* kbuf = (__bf16*)(ws + (size_t)M * H * 2);              // M*D
  __bf16* vbuf = (__bf16*)(ws + (size_t)(M * H + M * D) * 2);    // M*D
  __bf16* vtb  = (__bf16*)(ws + (size_t)(M * H + 2 * M * D) * 2);// Bsz*D*S
  __bf16* aob  = (__bf16*)(ws + (size_t)(M * H + 3 * M * D) * 2);// M*H

  dim3 blk(256);
  // Projections (fp32 A, bf16 out)
  gemm_wmma<float, __bf16><<<dim3(H / 64, M / 128), blk, 0, stream>>>(x, w_q, qbuf, M, H, H);
  gemm_wmma<float, __bf16><<<dim3(D / 64, M / 128), blk, 0, stream>>>(x, w_k, kbuf, M, D, H);
  gemm_wmma<float, __bf16><<<dim3(D / 64, M / 128), blk, 0, stream>>>(x, w_v, vbuf, M, D, H);
  // V transpose for contiguous P*V B-fragments
  transpose_v<<<(M * D) / 256, 256, 0, stream>>>(vbuf, vtb, S);
  // Streaming-softmax attention (TDM-staged K/V)
  mqa_attention<<<dim3(S / 64, HEADS, Bsz), dim3(128), 0, stream>>>(qbuf, kbuf, vtb, aob, S);
  // Output projection (bf16 A, fp32 out)
  gemm_wmma<__bf16, float><<<dim3(H / 64, M / 128), blk, 0, stream>>>(aob, w_out, out, M, H, H);
}
